// tensor_51548197486601
// MI455X (gfx1250) — matrix operand order repaired and checked by reading, not compiled, not run on hardware
//
#include <hip/hip_runtime.h>
#include <hip/hip_bf16.h>
#include <stdint.h>

// Problem constants (from reference)
#define B_TOT 1024
#define INND  512
#define HID   128
#define OUTD  64
#define INP1  513          // IN+1 (ones column prepended)
#define JP    544          // j-dim padded 513 -> 544 (17 chunks of K=32)
#define W1P_ROW ((size_t)INP1 * JP)   // 279072 bf16 per hidden row
#define MBLK  64           // batches per block
#define ISLICE 19          // i values per K-slice
#define NSLICE 27          // 27*19 = 513
#define MGRP  (B_TOT / MBLK)  // 16

typedef __attribute__((ext_vector_type(16))) __bf16 bf16x16;
typedef __attribute__((ext_vector_type(8)))  float  f32x8;

union APack { uint32_t u[8]; bf16x16 v; };
union BPack { int4 q[2];     bf16x16 v; };

__device__ __forceinline__ uint32_t pk_mul_bf16(uint32_t a, uint32_t b) {
  uint32_t d;
  asm volatile("v_pk_mul_bf16 %0, %1, %2" : "=v"(d) : "v"(a), "v"(b));
  return d;
}

// ---------------------------------------------------------------------------
// Kernel 1: convert W1 (fp32 [128, 513*513]) -> padded bf16 [128][513][544]
// ---------------------------------------------------------------------------
__global__ void w1_convert(const float* __restrict__ W1, __bf16* __restrict__ W1p) {
  const int i = blockIdx.x;   // 0..512
  const int h = blockIdx.y;   // 0..127
  const float*  src = W1  + (size_t)h * ((size_t)INP1 * INP1) + (size_t)i * INP1;
  __bf16*       dst = W1p + (size_t)h * W1P_ROW + (size_t)i * JP;
  for (int j = threadIdx.x; j < JP; j += blockDim.x) {
    float v = (j < INP1) ? src[j] : 0.0f;
    dst[j] = (__bf16)v;
  }
}

// ---------------------------------------------------------------------------
// Kernel 2: zero the y1 partial-sum buffer (split-K accumulates via atomics)
// ---------------------------------------------------------------------------
__global__ void zero_f32(float* __restrict__ p, int n) {
  int idx = blockIdx.x * blockDim.x + threadIdx.x;
  if (idx < n) p[idx] = 0.0f;
}

// ---------------------------------------------------------------------------
// Kernel 3: y1 += X^T * W1_h * T  via bf16 WMMA, fusion rows built on the fly.
// Block: 128 threads (4 waves). Each block: 64 batches x 128 hidden, one
// i-slice of 19. Each wave: 2 hidden tiles x 4 batch subtiles = 8 accums.
// ---------------------------------------------------------------------------
__global__ void __launch_bounds__(128)
bilinear_wmma(const float* __restrict__ img, const float* __restrict__ tab,
              const __bf16* __restrict__ W1p, float* __restrict__ y1) {
  extern __shared__ unsigned char smem[];
  __bf16* tls = (__bf16*)smem;                              // [MBLK][JP]
  __bf16* xls = (__bf16*)(smem + (size_t)MBLK * JP * 2);    // [MBLK][ISLICE]

  const int tid   = threadIdx.x;
  const int bbase = blockIdx.x * MBLK;
  const int ibase = blockIdx.y * ISLICE;

  // Stage t-tile into LDS (t = [1, tab_row], zero-padded to 544), bf16.
  for (int idx = tid; idx < MBLK * JP; idx += 128) {
    int r = idx / JP, j = idx - r * JP;
    float v;
    if (j == 0)          v = 1.0f;
    else if (j <= INND)  v = tab[(size_t)(bbase + r) * INND + (j - 1)];
    else                 v = 0.0f;
    tls[idx] = (__bf16)v;
  }
  // Stage x-slice into LDS (x = [1, img_row]).
  for (int idx = tid; idx < MBLK * ISLICE; idx += 128) {
    int r = idx / ISLICE, cc = idx - r * ISLICE;
    int i = ibase + cc;
    float v = (i == 0) ? 1.0f : img[(size_t)(bbase + r) * INND + (i - 1)];
    xls[idx] = (__bf16)v;
  }
  __syncthreads();

  const int wave = tid >> 5;
  const int lane = tid & 31;
  const int l15  = lane & 15;
  const int half = lane >> 4;

  // B-matrix (W1) row pointers: lane n holds column n (hidden unit), ISA
  // 16-bit B layout: lanes 0-15 hold K{0..7,16..23}, lanes 16-31 hold K{8..15,24..31}.
  const __bf16* pB0 = W1p + (size_t)(wave * 32 + l15)      * W1P_ROW;
  const __bf16* pB1 = W1p + (size_t)(wave * 32 + 16 + l15) * W1P_ROW;

  const f32x8 zero8 = {0.f, 0.f, 0.f, 0.f, 0.f, 0.f, 0.f, 0.f};
  f32x8 acc[4][2];
#pragma unroll
  for (int m = 0; m < 4; ++m) { acc[m][0] = zero8; acc[m][1] = zero8; }

  for (int ii = 0; ii < ISLICE; ++ii) {
    const int i = ibase + ii;
    // Per M-subtile broadcast scalar x[b,i], packed {x,x} for v_pk_mul_bf16.
    uint32_t xpk[4];
#pragma unroll
    for (int m = 0; m < 4; ++m) {
      __bf16 xv = xls[(m * 16 + l15) * ISLICE + ii];
      uint16_t bits; __builtin_memcpy(&bits, &xv, 2);
      xpk[m] = (uint32_t)bits * 0x00010001u;
    }
    const size_t koff = (size_t)i * JP;

    // 17 chunks of K=32; unroll x2 so ~16 WMMAs cover each batch of global
    // B loads (deeper pipelining window, amortized loop bookkeeping).
#pragma unroll 2
    for (int j0 = 0; j0 < JP; j0 += 32) {
      // Load two B tiles (W1), K{8*half..+7, 16+8*half..+7} per lane = 2x b128.
      BPack b0, b1;
      const __bf16* q0 = pB0 + koff + j0 + 8 * half;
      const __bf16* q1 = pB1 + koff + j0 + 8 * half;
      b0.q[0] = ((const int4*)q0)[0];
      b0.q[1] = *((const int4*)(q0 + 16));
      b1.q[0] = ((const int4*)q1)[0];
      b1.q[1] = *((const int4*)(q1 + 16));

#pragma unroll
      for (int m = 0; m < 4; ++m) {
        // Build A tile (fusion) on the fly: A[b,k] = x[b,i] * t[b,j0+k].
        // ISA 16-bit A layout: lanes 0-15 K{0..7,16..23}; lanes 16-31
        // K{8..15,24..31} -> two 16B LDS chunks at +8*half and +16+8*half.
        const __bf16* tr = tls + (m * 16 + l15) * JP + j0 + 8 * half;
        int4 t0 = ((const int4*)tr)[0];
        int4 t1 = *((const int4*)(tr + 16));
        const uint32_t* t0u = (const uint32_t*)&t0;
        const uint32_t* t1u = (const uint32_t*)&t1;
        APack a;
#pragma unroll
        for (int k = 0; k < 4; ++k) {
          a.u[k]     = pk_mul_bf16(t0u[k], xpk[m]);
          a.u[4 + k] = pk_mul_bf16(t1u[k], xpk[m]);
        }
        acc[m][0] = __builtin_amdgcn_wmma_f32_16x16x32_bf16(
            false, a.v, false, b0.v, (short)0, acc[m][0], false, false);
        acc[m][1] = __builtin_amdgcn_wmma_f32_16x16x32_bf16(
            false, a.v, false, b1.v, (short)0, acc[m][1], false, false);
      }
    }
  }

  // Split-K reduction: C/D layout lane = N (hidden), VGPR r -> M = r + 8*half.
#pragma unroll
  for (int m = 0; m < 4; ++m)
#pragma unroll
    for (int nt = 0; nt < 2; ++nt)
#pragma unroll
      for (int r = 0; r < 8; ++r) {
        int row = bbase + m * 16 + r + 8 * half;
        int col = wave * 32 + nt * 16 + l15;
        atomicAdd(&y1[(size_t)row * HID + col], acc[m][nt][r]);
      }
}

// ---------------------------------------------------------------------------
// Kernel 4: y2 = tanh((y1+b1) @ W2^T + b2); out = relu(y2 @ W3^T + b3). fp32.
// One block per batch row; trivially small (2.5e7 MACs total).
// ---------------------------------------------------------------------------
__global__ void __launch_bounds__(128)
mlp_tail(const float* __restrict__ y1, const float* __restrict__ b1,
         const float* __restrict__ W2, const float* __restrict__ b2,
         const float* __restrict__ W3, const float* __restrict__ b3,
         float* __restrict__ out) {
  __shared__ float s1[HID];
  __shared__ float s2[HID];
  const int b = blockIdx.x, t = threadIdx.x;
  s1[t] = y1[(size_t)b * HID + t] + b1[t];
  __syncthreads();
  float a = b2[t];
  const float* w = W2 + (size_t)t * HID;
#pragma unroll 8
  for (int k = 0; k < HID; ++k) a += s1[k] * w[k];
  s2[t] = tanhf(a);
  __syncthreads();
  if (t < OUTD) {
    float a3 = b3[t];
    const float* w3 = W3 + (size_t)t * HID;
#pragma unroll 8
    for (int k = 0; k < HID; ++k) a3 += s2[k] * w3[k];
    out[(size_t)b * OUTD + t] = fmaxf(a3, 0.0f);
  }
}

// ---------------------------------------------------------------------------
extern "C" void kernel_launch(void* const* d_in, const int* in_sizes, int n_in,
                              void* d_out, int out_size, void* d_ws, size_t ws_size,
                              hipStream_t stream) {
  (void)in_sizes; (void)n_in; (void)out_size; (void)ws_size;
  const float* img = (const float*)d_in[0];
  const float* tab = (const float*)d_in[1];
  const float* W1  = (const float*)d_in[2];
  const float* b1  = (const float*)d_in[3];
  const float* W2  = (const float*)d_in[4];
  const float* b2  = (const float*)d_in[5];
  const float* W3  = (const float*)d_in[6];
  const float* b3  = (const float*)d_in[7];
  float* out = (float*)d_out;

  // Workspace layout: [W1p bf16: 128*279072 = 71,442,432 B][y1 f32: 524,288 B]
  __bf16* W1p = (__bf16*)d_ws;
  float*  y1  = (float*)((char*)d_ws + (size_t)HID * W1P_ROW * sizeof(__bf16));

  dim3 gconv(INP1, HID);
  w1_convert<<<gconv, 256, 0, stream>>>(W1, W1p);

  zero_f32<<<(B_TOT * HID) / 256, 256, 0, stream>>>(y1, B_TOT * HID);

  dim3 gmain(MGRP, NSLICE);
  size_t shmem = (size_t)MBLK * JP * 2 + (size_t)MBLK * ISLICE * 2;  // 72,064 B
  bilinear_wmma<<<gmain, 128, shmem, stream>>>(img, tab, W1p, y1);

  mlp_tail<<<B_TOT, 128, 0, stream>>>(y1, b1, W2, b2, W3, b3, out);
}
